// GATConvBlock_35416300323761
// MI455X (gfx1250) — matrix operand order repaired and checked by reading, not compiled, not run on hardware
//
#include <hip/hip_runtime.h>
#include <stdint.h>

typedef __attribute__((ext_vector_type(16))) __bf16 v16bf;
typedef __attribute__((ext_vector_type(8)))  __bf16 v8bf;
typedef __attribute__((ext_vector_type(8)))  float  v8f;
typedef __attribute__((ext_vector_type(4)))  unsigned int u32x4;
typedef __attribute__((ext_vector_type(8)))  int  i32x8;
typedef __attribute__((ext_vector_type(4)))  int  i32x4;

#define HC        128   // H*C
#define HEADS     4
#define CH        32
#define NGROUP    16
#define NEG_SLOPE 0.2f

// ---------------------------------------------------------------- utilities
__global__ void k_fill_u32(uint32_t* __restrict__ p, size_t n, uint32_t v) {
  size_t i = (size_t)blockIdx.x * blockDim.x + threadIdx.x;
  if (i < n) p[i] = v;
}

// Native CDNA5 float atomic max (no return), device scope.
static __device__ __forceinline__ void atomic_max_f32(float* addr, float val) {
  asm volatile("global_atomic_max_num_f32 %0, %1, off scope:SCOPE_DEV"
               :: "v"((unsigned long long)(uintptr_t)addr), "v"(val)
               : "memory");
}

// -------------------------------------------- h = x @ W  via bf16 WMMA
// block = 256 threads (8 waves); each block: 128 rows of x, all 128 cols.
// x strip is DMA'd (TDM tensor_load_to_lds) into dynamic LDS as raw f32
// (overlapped with W staging), then converted to bf16.
// LDS: 32KB sX(bf16) + 32KB sWt(bf16) static + 64KB f32 stage dynamic = 128KB.
__global__ __launch_bounds__(256) void k_gemm(const float* __restrict__ x,
                                              const float* __restrict__ W,
                                              float* __restrict__ h, int n) {
  __shared__ __align__(16) __bf16 sX[128][128];   // [row][k]
  __shared__ __align__(16) __bf16 sWt[128][128];  // [col][k]  (transposed)
  extern __shared__ __align__(16) float sStage[]; // [128][128] f32, 64KB dynamic

  const int tid  = threadIdx.x;
  const int lane = tid & 31;
  const int wave = tid >> 5;
  const int row0 = blockIdx.x * 128;

  // ---- wave 0: issue one TDM descriptor for the whole 128x128 f32 x-strip.
  if (tid < 32) {
    unsigned lds_addr = (unsigned)(uintptr_t)&sStage[0];       // LDS byte offset
    unsigned long long ga = (unsigned long long)(uintptr_t)(x + (size_t)row0 * 128);
    unsigned rows = (unsigned)(n - row0);                      // OOB rows -> zero fill
    u32x4 g0 = { 1u,                                           // count=1, no gather
                 lds_addr,
                 (unsigned)(ga & 0xFFFFFFFFull),
                 (unsigned)((ga >> 32) & 0x01FFFFFFull) | 0x80000000u }; // type=2
    i32x8 g1 = { (int)0x00020000u,                 // data_size=4B, wg_mask=0
                 (int)(128u << 16),                // tensor_dim0 = 128
                 (int)((rows & 0xFFFFu) << 16),    // tensor_dim1 lo
                 (int)(((rows >> 16) & 0xFFFFu) | (128u << 16)), // dim1 hi | tile_dim0=128
                 128,                              // tile_dim1 = 128 (tile_dim2 = 0)
                 128,                              // tensor_dim0_stride = 128
                 0, 0 };                           // tensor_dim1_stride = 0 (2D)
    i32x4 gz4 = { 0, 0, 0, 0 };
    i32x8 gz8 = { 0, 0, 0, 0, 0, 0, 0, 0 };
    __builtin_amdgcn_tensor_load_to_lds(g0, g1, gz4, gz4, gz8, 0);
  }

  // ---- all waves: stage W transposed as bf16 while the TDM runs.
  for (int i = tid; i < 128 * 128; i += 256) {
    int k = i >> 7, c = i & 127;
    sWt[c][k] = (__bf16)W[i];                     // W is [k][c] row-major
  }

  if (tid < 32) __builtin_amdgcn_s_wait_tensorcnt(0);
  __syncthreads();

  // ---- convert staged f32 -> bf16
  for (int i = tid; i < 128 * 128; i += 256)
    sX[i >> 7][i & 127] = (__bf16)sStage[i];
  __syncthreads();

  const int m     = lane & 15;            // A: row M ; B: col N
  const int akoff = (lane >> 4) << 3;     // A lanes 0-15: K{0..7,16..23}; 16-31: K{8..15,24..31}
  const int bkoff = (lane >> 4) << 3;     // B lanes 0-15: K{0..7,16..23}; 16-31: K{8..15,24..31}
  const int rbase = wave << 4;

  v8f acc[8];
  const v8f vz = {0.f,0.f,0.f,0.f,0.f,0.f,0.f,0.f};
  #pragma unroll
  for (int nt = 0; nt < 8; ++nt) acc[nt] = vz;

  #pragma unroll
  for (int kb = 0; kb < 128; kb += 32) {
    v8bf alo = *(const v8bf*)&sX[rbase + m][kb + akoff];
    v8bf ahi = *(const v8bf*)&sX[rbase + m][kb + akoff + 16];
    v16bf a  = __builtin_shufflevector(alo, ahi, 0,1,2,3,4,5,6,7,8,9,10,11,12,13,14,15);
    // load ALL 8 B fragments first -> one dscnt wait, then 8 back-to-back WMMAs
    v16bf bf[8];
    #pragma unroll
    for (int nt = 0; nt < 8; ++nt) {
      v8bf blo = *(const v8bf*)&sWt[nt * 16 + m][kb + bkoff];
      v8bf bhi = *(const v8bf*)&sWt[nt * 16 + m][kb + bkoff + 16];
      bf[nt]   = __builtin_shufflevector(blo, bhi, 0,1,2,3,4,5,6,7,8,9,10,11,12,13,14,15);
    }
    #pragma unroll
    for (int nt = 0; nt < 8; ++nt)
      acc[nt] = __builtin_amdgcn_wmma_f32_16x16x32_bf16(
                    false, a, false, bf[nt], (short)0, acc[nt], false, false);
  }

  const int mr = (lane >> 4) << 3;        // D: VGPR r -> M = r + 8*(lane>=16)
  #pragma unroll
  for (int nt = 0; nt < 8; ++nt) {
    #pragma unroll
    for (int r = 0; r < 8; ++r) {
      int gr = row0 + rbase + mr + r;
      if (gr < n) h[(size_t)gr * 128 + nt * 16 + m] = acc[nt][r];
    }
  }
}

// ------------------------------------- per-node attention dots a_src, a_dst
__global__ void k_node_att(const float* __restrict__ h, const float* __restrict__ att_src,
                           const float* __restrict__ att_dst, float* __restrict__ a_src,
                           float* __restrict__ a_dst, int n) {
  int idx = blockIdx.x * blockDim.x + threadIdx.x;
  if (idx >= n * HEADS) return;
  int node = idx >> 2, hd = idx & 3;
  const float* hp = h + (size_t)node * HC + hd * CH;
  const float* as = att_src + hd * CH;
  const float* ad = att_dst + hd * CH;
  float s1 = 0.f, s2 = 0.f;
  #pragma unroll
  for (int c = 0; c < CH; ++c) { float v = hp[c]; s1 += v * as[c]; s2 += v * ad[c]; }
  a_src[idx] = s1; a_dst[idx] = s2;
}

// ------------------------------------- kedge[h] = dot(W_edge_h, att_edge_h)
__global__ void k_edge_coef(const float* __restrict__ W_edge,
                            const float* __restrict__ att_edge,
                            float* __restrict__ kedge) {
  __shared__ float red[HC];
  int t = threadIdx.x;
  red[t] = W_edge[t] * att_edge[t];
  __syncthreads();
  if (t < HEADS) {
    float s = 0.f;
    for (int c = 0; c < CH; ++c) s += red[t * CH + c];
    kedge[t] = s;
  }
}

// ------------------------------------- edge alpha + leaky relu + seg max
__global__ void k_edge_alpha(const float* __restrict__ a_src, const float* __restrict__ a_dst,
                             const float* __restrict__ edge_attr, const float* __restrict__ kedge,
                             const int* __restrict__ ei, float* __restrict__ alpha,
                             float* __restrict__ amax, int E) {
  int idx = blockIdx.x * blockDim.x + threadIdx.x;
  if (idx >= E * HEADS) return;
  int e = idx >> 2, hd = idx & 3;
  int s = ei[e], d = ei[E + e];
  float a = a_src[s * HEADS + hd] + a_dst[d * HEADS + hd] + edge_attr[e] * kedge[hd];
  a = (a >= 0.f) ? a : NEG_SLOPE * a;
  alpha[idx] = a;
  atomic_max_f32(&amax[d * HEADS + hd], a);
}

__global__ void k_fix_amax(float* __restrict__ amax, int n4) {
  int idx = blockIdx.x * blockDim.x + threadIdx.x;
  if (idx >= n4) return;
  float v = amax[idx];
  if (!isfinite(v)) amax[idx] = 0.f;
}

// ------------------------------------- ex = exp(alpha - amax[dst]); denom
__global__ void k_edge_exp(float* __restrict__ alpha, const float* __restrict__ amax,
                           float* __restrict__ denom, const int* __restrict__ ei, int E) {
  int idx = blockIdx.x * blockDim.x + threadIdx.x;
  if (idx >= E * HEADS) return;
  int e = idx >> 2, hd = idx & 3;
  int d = ei[E + e];
  float ex = __expf(alpha[idx] - amax[d * HEADS + hd]);
  alpha[idx] = ex;
  atomicAdd(&denom[d * HEADS + hd], ex);
}

// ------------------------------------- scatter: out[dst] += h[src]*att (32 lanes/edge)
__global__ __launch_bounds__(256) void k_scatter(const float* __restrict__ h,
                          const float* __restrict__ ex, const float* __restrict__ denom,
                          const int* __restrict__ ei, float* __restrict__ out, int E) {
  long long gid = (long long)blockIdx.x * blockDim.x + threadIdx.x;
  long long e = gid >> 5;
  if (e >= E) return;
  int lane = (int)(gid & 31);
  int f0 = lane << 2;                 // 4 floats per lane, 32 lanes = 128 features
  int hd = f0 >> 5;
  int s = ei[e], d = ei[(size_t)E + e];
  float att = ex[e * HEADS + hd] / (denom[(size_t)d * HEADS + hd] + 1e-16f);
  const float4 hv = *(const float4*)(h + (size_t)s * HC + f0);
  float* op = out + (size_t)d * HC + f0;
  atomicAdd(op + 0, hv.x * att);
  atomicAdd(op + 1, hv.y * att);
  atomicAdd(op + 2, hv.z * att);
  atomicAdd(op + 3, hv.w * att);
}

// ------------------------------------- GraphNorm
__global__ void k_counts(const int* __restrict__ bidx, float* __restrict__ counts, int n) {
  __shared__ float c[NGROUP];
  int t = threadIdx.x;
  if (t < NGROUP) c[t] = 0.f;
  __syncthreads();
  int node = blockIdx.x * blockDim.x + t;
  if (node < n) atomicAdd(&c[bidx[node]], 1.0f);
  __syncthreads();
  if (t < NGROUP) atomicAdd(&counts[t], c[t]);
}

// blockDim=128; thread f owns feature column f; 128 nodes per block.
__global__ __launch_bounds__(128) void k_gn_sum(const float* __restrict__ raw,
                        const float* __restrict__ bias, const int* __restrict__ bidx,
                        float* __restrict__ gsum, int n) {
  __shared__ float ls[NGROUP * HC];
  int f = threadIdx.x;
  #pragma unroll
  for (int g = 0; g < NGROUP; ++g) ls[g * HC + f] = 0.f;
  int nb = blockIdx.x * 128;
  float bf = bias[f];
  for (int r = 0; r < 128; ++r) {
    int node = nb + r;
    if (node >= n) break;
    ls[bidx[node] * HC + f] += raw[(size_t)node * HC + f] + bf;
  }
  #pragma unroll
  for (int g = 0; g < NGROUP; ++g) atomicAdd(&gsum[g * HC + f], ls[g * HC + f]);
}

__global__ void k_gn_mean(float* __restrict__ gsum, const float* __restrict__ counts) {
  int idx = blockIdx.x * blockDim.x + threadIdx.x;
  if (idx >= NGROUP * HC) return;
  gsum[idx] /= counts[idx >> 7];
}

__global__ __launch_bounds__(128) void k_gn_center(const float* __restrict__ raw,
    const float* __restrict__ bias, const float* __restrict__ mean,
    const float* __restrict__ mscale, const int* __restrict__ bidx,
    float* __restrict__ outc, float* __restrict__ gvar, int n) {
  __shared__ float lv[NGROUP * HC];
  int f = threadIdx.x;
  #pragma unroll
  for (int g = 0; g < NGROUP; ++g) lv[g * HC + f] = 0.f;
  int nb = blockIdx.x * 128;
  float bf = bias[f], ms = mscale[f];
  for (int r = 0; r < 128; ++r) {
    int node = nb + r;
    if (node >= n) break;
    int g = bidx[node];
    float cen = raw[(size_t)node * HC + f] + bf - ms * mean[g * HC + f];
    outc[(size_t)node * HC + f] = cen;
    lv[g * HC + f] += cen * cen;
  }
  #pragma unroll
  for (int g = 0; g < NGROUP; ++g) atomicAdd(&gvar[g * HC + f], lv[g * HC + f]);
}

__global__ void k_gn_invstd(float* __restrict__ gvar, const float* __restrict__ counts) {
  int idx = blockIdx.x * blockDim.x + threadIdx.x;
  if (idx >= NGROUP * HC) return;
  gvar[idx] = rsqrtf(gvar[idx] / counts[idx >> 7] + 1e-5f);
}

__global__ void k_gn_final(float* __restrict__ out, const float* __restrict__ gnw,
                           const float* __restrict__ gnb, const float* __restrict__ inv,
                           const int* __restrict__ bidx, int n) {
  long long idx = (long long)blockIdx.x * blockDim.x + threadIdx.x;
  if (idx >= (long long)n * HC) return;
  int f = (int)(idx & 127);
  int node = (int)(idx >> 7);
  int g = bidx[node];
  out[idx] = gnw[f] * out[idx] * inv[g * HC + f] + gnb[f];
}

// ---------------------------------------------------------------- launcher
extern "C" void kernel_launch(void* const* d_in, const int* in_sizes, int n_in,
                              void* d_out, int out_size, void* d_ws, size_t ws_size,
                              hipStream_t stream) {
  const float* x         = (const float*)d_in[0];
  const float* edge_attr = (const float*)d_in[1];
  const float* W         = (const float*)d_in[2];
  const float* att_src   = (const float*)d_in[3];
  const float* att_dst   = (const float*)d_in[4];
  const float* W_edge    = (const float*)d_in[5];
  const float* att_edge  = (const float*)d_in[6];
  const float* bias      = (const float*)d_in[7];
  const float* gn_weight = (const float*)d_in[8];
  const float* gn_bias   = (const float*)d_in[9];
  const float* gn_mscale = (const float*)d_in[10];
  const int*   edge_index  = (const int*)d_in[11];
  const int*   batch_index = (const int*)d_in[12];

  const int N = in_sizes[0] / HC;   // IN == H*C == 128
  const int E = in_sizes[1];
  (void)n_in; (void)out_size; (void)ws_size;

  char* wsb = (char*)d_ws;
  size_t off = 0;
  auto alloc = [&](size_t bytes) -> void* {
    void* p = wsb + off;
    off += (bytes + 255) & ~(size_t)255;
    return p;
  };
  float* h      = (float*)alloc((size_t)N * HC * 4);
  float* a_src  = (float*)alloc((size_t)N * HEADS * 4);
  float* a_dst  = (float*)alloc((size_t)N * HEADS * 4);
  float* kedge  = (float*)alloc(256);
  float* alpha  = (float*)alloc((size_t)E * HEADS * 4);
  float* amax   = (float*)alloc((size_t)N * HEADS * 4);
  float* denom  = (float*)alloc((size_t)N * HEADS * 4);
  float* raw    = (float*)alloc((size_t)N * HC * 4);
  float* gsum   = (float*)alloc(NGROUP * HC * 4);
  float* gvar   = (float*)alloc(NGROUP * HC * 4);
  float* counts = (float*)alloc(256);
  float* outp   = (float*)d_out;

  auto nb = [](long long t, int b) { return (int)((t + b - 1) / b); };

  // init accumulators (harness does not re-zero between replays)
  k_fill_u32<<<nb((long long)N * HEADS, 256), 256, 0, stream>>>((uint32_t*)amax,  (size_t)N * HEADS, 0xFF800000u); // -inf
  k_fill_u32<<<nb((long long)N * HEADS, 256), 256, 0, stream>>>((uint32_t*)denom, (size_t)N * HEADS, 0u);
  k_fill_u32<<<nb((long long)N * HC,    256), 256, 0, stream>>>((uint32_t*)raw,   (size_t)N * HC,    0u);
  k_fill_u32<<<nb(NGROUP * HC, 256), 256, 0, stream>>>((uint32_t*)gsum, (size_t)NGROUP * HC, 0u);
  k_fill_u32<<<nb(NGROUP * HC, 256), 256, 0, stream>>>((uint32_t*)gvar, (size_t)NGROUP * HC, 0u);
  k_fill_u32<<<1, 256, 0, stream>>>((uint32_t*)counts, (size_t)NGROUP, 0u);

  // GAT
  k_gemm<<<nb(N, 128), 256, 65536, stream>>>(x, W, h, N);   // 64KB dynamic LDS stage
  k_node_att<<<nb((long long)N * HEADS, 256), 256, 0, stream>>>(h, att_src, att_dst, a_src, a_dst, N);
  k_edge_coef<<<1, HC, 0, stream>>>(W_edge, att_edge, kedge);
  k_edge_alpha<<<nb((long long)E * HEADS, 256), 256, 0, stream>>>(a_src, a_dst, edge_attr, kedge,
                                                                  edge_index, alpha, amax, E);
  k_fix_amax<<<nb((long long)N * HEADS, 256), 256, 0, stream>>>(amax, N * HEADS);
  k_edge_exp<<<nb((long long)E * HEADS, 256), 256, 0, stream>>>(alpha, amax, denom, edge_index, E);
  k_scatter<<<nb((long long)E * 32, 256), 256, 0, stream>>>(h, alpha, denom, edge_index, raw, E);

  // GraphNorm
  k_gn_sum<<<nb(N, 128), 128, 0, stream>>>(raw, bias, batch_index, gsum, N);
  k_counts<<<nb(N, 256), 256, 0, stream>>>(batch_index, counts, N);
  k_gn_mean<<<nb(NGROUP * HC, 256), 256, 0, stream>>>(gsum, counts);
  k_gn_center<<<nb(N, 128), 128, 0, stream>>>(raw, bias, gsum, gn_mscale, batch_index, outp, gvar, N);
  k_gn_invstd<<<nb(NGROUP * HC, 256), 256, 0, stream>>>(gvar, counts);
  k_gn_final<<<nb((long long)N * HC, 256), 256, 0, stream>>>(outp, gn_weight, gn_bias, gvar, batch_index, N);
}
